// SimpleDecoder_11390253269015
// MI455X (gfx1250) — matrix operand order repaired and checked by reading, not compiled, not run on hardware
//
#include <hip/hip_runtime.h>
#include <hip/hip_bf16.h>
#include <stdint.h>

// Problem constants (from reference)
#define S_   1024
#define B_   32
#define DIN_ 1024
#define E_   128
#define H_   1024
#define V_   5000
#define VPAD_ 5120          // V rounded up to 128 (zero-padded W_out rows)
#define DT_  (DIN_ + E_)    // 1152
#define ROWS_ (S_ * B_)     // 32768

typedef unsigned short ushort_t;
typedef __bf16 bf16_t;
typedef bf16_t v16bf __attribute__((ext_vector_type(16)));
typedef float  v8f   __attribute__((ext_vector_type(8)));

// ---------------------------------------------------------------- helpers ---

__device__ inline ushort_t f2bf(float f) {
  union { float f; uint32_t u; } cv; cv.f = f;
  const uint32_t u = cv.u;
  const uint32_t r = 0x7FFFu + ((u >> 16) & 1u);   // round-to-nearest-even
  return (ushort_t)((u + r) >> 16);
}

union FragAB {        // 16 bf16 = 32 bytes, loaded as two 16B chunks
  v16bf v;
  uint4 q[2];
};

__device__ inline v8f wmma_bf16(const FragAB& a, const FragAB& b, v8f c) {
  // D = A(16x32 bf16) * B(32x16 bf16) + C(16x16 f32)
  return __builtin_amdgcn_wmma_f32_16x16x32_bf16(
      false, a.v, false, b.v, (short)0, c, false, false);
}

// CDNA5 async global->LDS copy (16B per lane), tracked by ASYNCcnt.
// ISA 10.2: for pointers in the LDS aperture, only the low address dword is used,
// so the low 32 bits of a generic __shared__ pointer are the LDS byte address.
__device__ inline void async_ld16(void* lds_dst, const void* gsrc) {
  const uint32_t loff = (uint32_t)(uintptr_t)lds_dst;
  const uint64_t gaddr = (uint64_t)(uintptr_t)gsrc;
  asm volatile("global_load_async_to_lds_b128 %0, %1, off"
               :: "v"(loff), "v"(gaddr) : "memory");
}
__device__ inline void wait_async_le4() {
  asm volatile("s_wait_asynccnt 0x4" ::: "memory");
}
__device__ inline void wait_async_0() {
  asm volatile("s_wait_asynccnt 0x0" ::: "memory");
}

__device__ inline float block_reduce_sum(float v, float* red) {
  const int tid = threadIdx.x;
  red[tid] = v; __syncthreads();
  for (int s = 128; s > 0; s >>= 1) {
    if (tid < s) red[tid] += red[tid + s];
    __syncthreads();
  }
  const float r = red[0]; __syncthreads();
  return r;
}

__device__ inline float block_reduce_max(float v, float* red) {
  const int tid = threadIdx.x;
  red[tid] = v; __syncthreads();
  for (int s = 128; s > 0; s >>= 1) {
    if (tid < s) red[tid] = fmaxf(red[tid], red[tid + s]);
    __syncthreads();
  }
  const float r = red[0]; __syncthreads();
  return r;
}

// ------------------------------------------------------ generic WMMA GEMM ---
// C[M,N](f32) = A_bf16[M,K] @ W_bf16[N,K]^T (+bias) (relu?) (+residual)
// Block tile 128x128, BK=32. 8 waves arranged 2(M)x4(N); each wave owns
// 64x32 = 4x2 WMMA tiles.  Double-buffered LDS fed by async global->LDS
// loads: stage k+1 is in flight (ASYNCcnt<=4) while stage k runs its WMMAs.
// Last iteration is peeled so the steady-state loop is branch-free.
// Requirements: M % 128 == 0; K % 32 == 0; W has gridDim.x*128 rows (caller
// zero-pads); stores are bounds-checked against N.
__global__ __launch_bounds__(256) void gemm_bf16_kernel(
    const ushort_t* __restrict__ A, const ushort_t* __restrict__ W,
    const float* __restrict__ bias, const float* __restrict__ residual,
    float* __restrict__ Cf, ushort_t* __restrict__ Cb,
    int M, int N, int K, int relu)
{
  __shared__ __align__(16) ushort_t As[2][128][32];   // 16 KB
  __shared__ __align__(16) ushort_t Bs[2][128][32];   // 16 KB

  const int tid  = threadIdx.x;
  const int wave = tid >> 5;
  const int lane = tid & 31;
  const int l15  = lane & 15;
  const int lhi  = lane >> 4;
  const int wm   = wave >> 2;          // 0..1  (64 rows each)
  const int wn   = wave & 3;           // 0..3  (32 cols each)
  const int bm   = blockIdx.y * 128;
  const int bn   = blockIdx.x * 128;

  v8f acc[4][2];
  #pragma unroll
  for (int i = 0; i < 4; ++i)
    #pragma unroll
    for (int j = 0; j < 2; ++j)
      #pragma unroll
      for (int e = 0; e < 8; ++e) acc[i][j][e] = 0.f;

  const int ldsrow = tid >> 1;               // 0..127
  const int ldscol = (tid & 1) << 4;         // 0 / 16

  // one staging = 4 async b128 instructions per wave (2 for A, 2 for B)
  auto stage = [&](int buf, const ushort_t* pa, const ushort_t* pb) {
    async_ld16(&As[buf][ldsrow][ldscol],     pa);
    async_ld16(&As[buf][ldsrow][ldscol + 8], pa + 8);
    async_ld16(&Bs[buf][ldsrow][ldscol],     pb);
    async_ld16(&Bs[buf][ldsrow][ldscol + 8], pb + 8);
  };

  // 8 WMMA tiles per K-stage, fragments per documented CDNA5 wave32 layouts
  auto compute = [&](int buf) {
    FragAB a[4], b[2];
    #pragma unroll
    for (int mt = 0; mt < 4; ++mt) {
      const int ar = wm * 64 + mt * 16 + l15;    // A row = lane[3:0]
      const int kb = lhi * 8;                    // lanes 16-31 hold K+8 / K+24
      a[mt].q[0] = *(const uint4*)&As[buf][ar][kb];
      a[mt].q[1] = *(const uint4*)&As[buf][ar][kb + 16];
    }
    #pragma unroll
    for (int nt = 0; nt < 2; ++nt) {
      const int br = wn * 32 + nt * 16 + l15;    // B col = lane[3:0]
      b[nt].q[0] = *(const uint4*)&Bs[buf][br][lhi * 8];  // hi lanes: K+8 / K+24
      b[nt].q[1] = *(const uint4*)&Bs[buf][br][lhi * 8 + 16];
    }
    #pragma unroll
    for (int mt = 0; mt < 4; ++mt)
      #pragma unroll
      for (int nt = 0; nt < 2; ++nt)
        acc[mt][nt] = wmma_bf16(a[mt], b[nt], acc[mt][nt]);
  };

  const ushort_t* pA = A + (size_t)(bm + ldsrow) * (size_t)K + ldscol;
  const ushort_t* pB = W + (size_t)(bn + ldsrow) * (size_t)K + ldscol;

  const int nk = K >> 5;
  stage(0, pA, pB);
  pA += 32; pB += 32;

  // steady state: branch-free (stage next / wait<=4 / barrier / wmma / barrier)
  for (int i = 0; i < nk - 1; ++i) {
    const int buf = i & 1;
    stage(buf ^ 1, pA, pB);           // prefetch next stage into other buffer
    pA += 32; pB += 32;
    wait_async_le4();                 // stage i complete (in-order ASYNCcnt)
    __syncthreads();                  // all waves' stage-i data visible
    compute(buf);
    __syncthreads();                  // done reading buf before it refills
  }
  // peeled last stage
  wait_async_0();
  __syncthreads();
  compute((nk - 1) & 1);

  // Epilogue. C/D layout: element (vgpr r, lane l) -> M = r + 8*l[4], N = l[3:0]
  #pragma unroll
  for (int mt = 0; mt < 4; ++mt) {
    #pragma unroll
    for (int nt = 0; nt < 2; ++nt) {
      const int col = bn + wn * 32 + nt * 16 + l15;
      if (col >= N) continue;
      const float bv = bias ? bias[col] : 0.f;
      #pragma unroll
      for (int r = 0; r < 8; ++r) {
        const int row = bm + wm * 64 + mt * 16 + r + lhi * 8;
        float v = acc[mt][nt][r] + bv;
        if (relu) v = v > 0.f ? v : 0.f;
        if (residual) v += residual[(size_t)row * N + col];
        if (Cf) Cf[(size_t)row * N + col] = v;
        if (Cb) Cb[(size_t)row * N + col] = f2bf(v);
      }
    }
  }
}

// ----------------------------------------------------------- LSTM step ------
// One launch per timestep. Grid = H/128 blocks; each block owns hidden columns
// [nb, nb+128) for ALL four gates, so the full gate vector for those columns is
// local -> fuse sigmoid/tanh/c/h update. h ping-pongs (bf16) to avoid races.
// The Whh B-fragments are software-pipelined in registers (two sets) so the
// global loads for K-stage k+32 overlap the WMMAs of stage k.
__global__ __launch_bounds__(256) void lstm_step_kernel(
    const ushort_t* __restrict__ Whh,   // [4H, H] bf16
    const float*    __restrict__ gx_t,  // [B, 4H] = x_t@Wih^T + bih
    const float*    __restrict__ bhh,   // [4H]
    const ushort_t* __restrict__ h_in,  // [B, H] bf16
    ushort_t*       __restrict__ h_out, // [B, H] bf16
    float*          __restrict__ c,     // [B, H] f32 (column-partitioned)
    float*          __restrict__ hs_out)// [B, H] f32 -> d_out slice
{
  extern __shared__ __align__(16) char smem[];
  ushort_t* Hs = (ushort_t*)smem;                     // [32][1024] = 64 KB
  float*    Gs = (float*)(smem + 32 * 1024 * 2);      // [4][32][128] = 64 KB

  const int tid  = threadIdx.x;
  const int wave = tid >> 5;
  const int lane = tid & 31;
  const int l15  = lane & 15;
  const int lhi  = lane >> 4;
  const int gate = wave >> 1;      // 0..3 (i,f,g,o)
  const int mh   = wave & 1;       // rows 0..15 / 16..31
  const int nb   = blockIdx.x * 128;

  // stage full h (32x1024 bf16 = 64 KB) via async global->LDS
  for (int i = tid; i < (32 * 1024) / 8; i += 256)
    async_ld16(&Hs[i * 8], h_in + i * 8);

  v8f acc[8];
  #pragma unroll
  for (int nt = 0; nt < 8; ++nt)
    #pragma unroll
    for (int e = 0; e < 8; ++e) acc[nt][e] = 0.f;

  // B-fragment loader (8 column tiles for this wave's gate / row-half)
  auto load_b = [&](FragAB* bf, int k0) {
    #pragma unroll
    for (int nt = 0; nt < 8; ++nt) {
      const size_t wr = (size_t)(gate * H_ + nb + nt * 16 + l15) * (size_t)H_;
      bf[nt].q[0] = *(const uint4*)&Whh[wr + k0 + lhi * 8];
      bf[nt].q[1] = *(const uint4*)&Whh[wr + k0 + lhi * 8 + 16];
    }
  };
  auto do_k = [&](int k0, FragAB* bf) {
    FragAB a;
    const int ar = mh * 16 + l15;
    a.q[0] = *(const uint4*)&Hs[ar * 1024 + k0 + lhi * 8];
    a.q[1] = *(const uint4*)&Hs[ar * 1024 + k0 + lhi * 8 + 16];
    #pragma unroll
    for (int nt = 0; nt < 8; ++nt) acc[nt] = wmma_bf16(a, bf[nt], acc[nt]);
  };

  FragAB b0[8], b1[8];
  load_b(b0, 0);             // overlaps with the async h staging above
  wait_async_0();            // h tile resident in LDS
  __syncthreads();

  for (int k0 = 0; k0 < H_; k0 += 64) {
    load_b(b1, k0 + 32);                 // prefetch stage k+32
    do_k(k0, b0);
    if (k0 + 64 < H_) load_b(b0, k0 + 64);  // prefetch stage k+64
    do_k(k0 + 32, b1);
  }

  // deposit partial gate pre-activations
  #pragma unroll
  for (int nt = 0; nt < 8; ++nt)
    #pragma unroll
    for (int r = 0; r < 8; ++r) {
      const int row  = mh * 16 + r + lhi * 8;
      const int colL = nt * 16 + l15;
      Gs[(gate * 32 + row) * 128 + colL] = acc[nt][r];
    }
  __syncthreads();

  // fused gates + state update for our 32x128 slice
  for (int i = tid; i < 32 * 128; i += 256) {
    const int b  = i >> 7;
    const int cl = i & 127;
    const int hc = nb + cl;
    const float* gxr = gx_t + (size_t)b * (4 * H_);
    const float gi = Gs[(0 * 32 + b) * 128 + cl] + gxr[0 * H_ + hc] + bhh[0 * H_ + hc];
    const float gf = Gs[(1 * 32 + b) * 128 + cl] + gxr[1 * H_ + hc] + bhh[1 * H_ + hc];
    const float gg = Gs[(2 * 32 + b) * 128 + cl] + gxr[2 * H_ + hc] + bhh[2 * H_ + hc];
    const float go = Gs[(3 * 32 + b) * 128 + cl] + gxr[3 * H_ + hc] + bhh[3 * H_ + hc];
    const float si = 1.f / (1.f + __expf(-gi));
    const float sf = 1.f / (1.f + __expf(-gf));
    const float so = 1.f / (1.f + __expf(-go));
    const size_t idx = (size_t)b * H_ + hc;
    const float cv = sf * c[idx] + si * tanhf(gg);
    const float hv = so * tanhf(cv);
    c[idx]      = cv;
    h_out[idx]  = f2bf(hv);
    hs_out[idx] = hv;
  }
}

// --------------------------------------------------------- LN-ish kernels ---

// LN over concat(x_row, emb[label]) -> bf16 (WMMA GEMM1 input)
__global__ __launch_bounds__(256) void ln_in_kernel(
    const float* __restrict__ x, const int* __restrict__ labels,
    const float* __restrict__ emb, const float* __restrict__ g,
    const float* __restrict__ bb, ushort_t* __restrict__ out_bf)
{
  __shared__ float rowbuf[DT_];
  __shared__ float red[256];
  const int r   = blockIdx.x;
  const int tid = threadIdx.x;
  const float* xr = x + (size_t)r * DIN_;
  for (int cidx = tid; cidx < DIN_; cidx += 256) rowbuf[cidx] = xr[cidx];
  const int lab = labels[r];
  const float* er = emb + (size_t)lab * E_;
  for (int cidx = tid; cidx < E_; cidx += 256) rowbuf[DIN_ + cidx] = er[cidx];
  __syncthreads();

  float s = 0.f, s2 = 0.f;
  for (int cidx = tid; cidx < DT_; cidx += 256) {
    const float v = rowbuf[cidx]; s += v; s2 += v * v;
  }
  s  = block_reduce_sum(s, red);
  s2 = block_reduce_sum(s2, red);
  const float mean = s / DT_;
  const float rstd = rsqrtf(s2 / DT_ - mean * mean + 1e-5f);
  for (int cidx = tid; cidx < DT_; cidx += 256)
    out_bf[(size_t)r * DT_ + cidx] =
        f2bf((rowbuf[cidx] - mean) * rstd * g[cidx] + bb[cidx]);
}

// Generic row LayerNorm. v = X (+Y).  Optionally emit pre-LN sum (f32),
// post-LN f32 and/or post-LN bf16.
__global__ __launch_bounds__(256) void rowln_kernel(
    const float* __restrict__ X, const float* __restrict__ Y,
    const float* __restrict__ g, const float* __restrict__ bb,
    float* __restrict__ out_pre, float* __restrict__ out_f32,
    ushort_t* __restrict__ out_bf, int D)
{
  __shared__ float red[256];
  const size_t base = (size_t)blockIdx.x * (size_t)D;
  const int tid = threadIdx.x;
  float s = 0.f, s2 = 0.f;
  for (int cidx = tid; cidx < D; cidx += 256) {
    float v = X[base + cidx];
    if (Y) v += Y[base + cidx];
    if (out_pre) out_pre[base + cidx] = v;
    s += v; s2 += v * v;
  }
  s  = block_reduce_sum(s, red);
  s2 = block_reduce_sum(s2, red);
  const float mean = s / D;
  const float rstd = rsqrtf(s2 / D - mean * mean + 1e-5f);
  for (int cidx = tid; cidx < D; cidx += 256) {
    float v = X[base + cidx];
    if (Y) v += Y[base + cidx];
    const float y = (v - mean) * rstd * g[cidx] + bb[cidx];
    if (out_f32) out_f32[base + cidx] = y;
    if (out_bf)  out_bf[base + cidx]  = f2bf(y);
  }
}

// In-place LayerNorm + log_softmax over V per row
__global__ __launch_bounds__(256) void ln_logsoftmax_kernel(
    float* __restrict__ X, const float* __restrict__ g,
    const float* __restrict__ bb, int D)
{
  __shared__ float red[256];
  float* row = X + (size_t)blockIdx.x * (size_t)D;
  const int tid = threadIdx.x;
  float s = 0.f, s2 = 0.f;
  for (int cidx = tid; cidx < D; cidx += 256) {
    const float v = row[cidx]; s += v; s2 += v * v;
  }
  s  = block_reduce_sum(s, red);
  s2 = block_reduce_sum(s2, red);
  const float mean = s / D;
  const float rstd = rsqrtf(s2 / D - mean * mean + 1e-5f);

  float mx = -3.402823466e38f;
  for (int cidx = tid; cidx < D; cidx += 256) {
    const float y = (row[cidx] - mean) * rstd * g[cidx] + bb[cidx];
    row[cidx] = y;
    mx = fmaxf(mx, y);
  }
  mx = block_reduce_max(mx, red);
  float es = 0.f;
  for (int cidx = tid; cidx < D; cidx += 256) es += __expf(row[cidx] - mx);
  es = block_reduce_sum(es, red);
  const float ls = mx + __logf(es);
  for (int cidx = tid; cidx < D; cidx += 256) row[cidx] -= ls;
}

// ----------------------------------------------------------- tiny kernels ---

__global__ void cvt_bf16_kernel(const float* __restrict__ src,
                                ushort_t* __restrict__ dst, size_t n) {
  const size_t i = (size_t)blockIdx.x * blockDim.x + threadIdx.x;
  if (i < n) dst[i] = f2bf(src[i]);
}

__global__ void zero_bf16_kernel(ushort_t* __restrict__ dst, size_t n) {
  const size_t i = (size_t)blockIdx.x * blockDim.x + threadIdx.x;
  if (i < n) dst[i] = 0;
}

__global__ void init_state_kernel(ushort_t* __restrict__ h0,
                                  float* __restrict__ c0, int n) {
  const int i = blockIdx.x * blockDim.x + threadIdx.x;
  if (i < n) { h0[i] = 0; c0[i] = 0.f; }
}

// ------------------------------------------------------------------ host ----

extern "C" void kernel_launch(void* const* d_in, const int* in_sizes, int n_in,
                              void* d_out, int out_size, void* d_ws, size_t ws_size,
                              hipStream_t stream)
{
  (void)in_sizes; (void)n_in; (void)out_size; (void)ws_size;

  const float* x        = (const float*)d_in[0];
  const int*   labels   = (const int*)  d_in[1];
  const float* emb      = (const float*)d_in[2];
  const float* ln_in_g  = (const float*)d_in[3];
  const float* ln_in_b  = (const float*)d_in[4];
  const float* W_hs     = (const float*)d_in[5];
  const float* b_hs     = (const float*)d_in[6];
  const float* ln_rnn_g = (const float*)d_in[7];
  const float* ln_rnn_b = (const float*)d_in[8];
  const float* Wih      = (const float*)d_in[9];
  const float* Whh      = (const float*)d_in[10];
  const float* bih      = (const float*)d_in[11];
  const float* bhh      = (const float*)d_in[12];
  const float* ln_mlp_g = (const float*)d_in[13];
  const float* ln_mlp_b = (const float*)d_in[14];
  const float* W_m1     = (const float*)d_in[15];
  const float* b_m1     = (const float*)d_in[16];
  const float* W_m2     = (const float*)d_in[17];
  const float* b_m2     = (const float*)d_in[18];
  const float* W_out    = (const float*)d_in[19];
  const float* b_out    = (const float*)d_in[20];
  const float* ln_out_g = (const float*)d_in[21];
  const float* ln_out_b = (const float*)d_in[22];

  // -------- workspace carve-up (all offsets 256B aligned) --------
  char* ws = (char*)d_ws;
  size_t off = 0;
  auto take = [&](size_t bytes) -> char* {
    char* p = ws + off;
    off = (off + bytes + 255) & ~(size_t)255;
    return p;
  };
  ushort_t* whs_bf  = (ushort_t*)take((size_t)H_ * DT_ * 2);
  ushort_t* wih_bf  = (ushort_t*)take((size_t)4 * H_ * H_ * 2);
  ushort_t* whh_bf  = (ushort_t*)take((size_t)4 * H_ * H_ * 2);
  ushort_t* wm1_bf  = (ushort_t*)take((size_t)H_ * H_ * 2);
  ushort_t* wm2_bf  = (ushort_t*)take((size_t)H_ * H_ * 2);
  ushort_t* wout_bf = (ushort_t*)take((size_t)VPAD_ * H_ * 2);  // zero-padded
  ushort_t* bwtot_bf= (ushort_t*)take((size_t)ROWS_ * DT_ * 2);
  float*    pre1    = (float*)   take((size_t)ROWS_ * H_ * 4);
  float*    rnn_f32 = (float*)   take((size_t)ROWS_ * H_ * 4);
  ushort_t* rnn_bf  = (ushort_t*)take((size_t)ROWS_ * H_ * 2);
  float*    gx      = (float*)   take((size_t)ROWS_ * 4 * H_ * 4);
  float*    mlp_in  = (float*)   take((size_t)ROWS_ * H_ * 4);
  ushort_t* z_bf    = (ushort_t*)take((size_t)ROWS_ * H_ * 2);
  ushort_t* t1_bf   = (ushort_t*)take((size_t)ROWS_ * H_ * 2);
  ushort_t* fin_bf  = (ushort_t*)take((size_t)ROWS_ * H_ * 2);
  ushort_t* hping   = (ushort_t*)take((size_t)2 * B_ * H_ * 2);
  float*    cbuf    = (float*)   take((size_t)B_ * H_ * 4);

  float* hs_out = (float*)d_out;                          // [S,B,H]
  float* scores = (float*)d_out + (size_t)ROWS_ * H_;     // [S,B,V]

  // -------- weight conversions to bf16 (cheap, every call: deterministic) ---
  auto cvt = [&](const float* s, ushort_t* dptr, size_t n) {
    cvt_bf16_kernel<<<dim3((unsigned)((n + 255) / 256)), dim3(256), 0, stream>>>(s, dptr, n);
  };
  cvt(W_hs,  whs_bf,  (size_t)H_ * DT_);
  cvt(Wih,   wih_bf,  (size_t)4 * H_ * H_);
  cvt(Whh,   whh_bf,  (size_t)4 * H_ * H_);
  cvt(W_m1,  wm1_bf,  (size_t)H_ * H_);
  cvt(W_m2,  wm2_bf,  (size_t)H_ * H_);
  cvt(W_out, wout_bf, (size_t)V_ * H_);
  { // zero the padded tail rows of W_out (rows V_..VPAD_-1)
    const size_t npad = (size_t)(VPAD_ - V_) * H_;
    zero_bf16_kernel<<<dim3((unsigned)((npad + 255) / 256)), dim3(256), 0, stream>>>(
        wout_bf + (size_t)V_ * H_, npad);
  }

  // 1) input LN over concat(x, emb[labels]) -> bf16
  ln_in_kernel<<<ROWS_, 256, 0, stream>>>(x, labels, emb, ln_in_g, ln_in_b, bwtot_bf);

  // 2) rnn_pre = bw_total @ W_hs^T + b_hs  (WMMA)
  gemm_bf16_kernel<<<dim3(H_ / 128, ROWS_ / 128), 256, 0, stream>>>(
      bwtot_bf, whs_bf, b_hs, nullptr, pre1, nullptr, ROWS_, H_, DT_, 0);

  // 3) rnn_in = LN(rnn_pre) -> f32 (residual) + bf16 (GEMM input)
  rowln_kernel<<<ROWS_, 256, 0, stream>>>(pre1, nullptr, ln_rnn_g, ln_rnn_b,
                                          nullptr, rnn_f32, rnn_bf, H_);

  // 4) gx[t] = rnn_in @ Wih^T + bih for ALL timesteps in one big WMMA GEMM
  gemm_bf16_kernel<<<dim3(4 * H_ / 128, ROWS_ / 128), 256, 0, stream>>>(
      rnn_bf, wih_bf, bih, nullptr, gx, nullptr, ROWS_, 4 * H_, H_, 0);

  // 5) sequential LSTM recurrence (h@Whh^T via WMMA, fused gates)
  init_state_kernel<<<(B_ * H_ + 255) / 256, 256, 0, stream>>>(hping, cbuf, B_ * H_);
  const size_t lstm_lds = (size_t)32 * 1024 * 2 + (size_t)4 * 32 * 128 * 4; // 128 KB
  for (int t = 0; t < S_; ++t) {
    const ushort_t* hin  = hping + (size_t)(t & 1) * B_ * H_;
    ushort_t*       hout = hping + (size_t)((t + 1) & 1) * B_ * H_;
    lstm_step_kernel<<<H_ / 128, 256, lstm_lds, stream>>>(
        whh_bf, gx + (size_t)t * B_ * 4 * H_, bhh, hin, hout, cbuf,
        hs_out + (size_t)t * B_ * H_);
  }

  // 6) mlp_in = rnn_in + hs (pre-LN, f32) ; z = LN(mlp_in) -> bf16
  rowln_kernel<<<ROWS_, 256, 0, stream>>>(rnn_f32, hs_out, ln_mlp_g, ln_mlp_b,
                                          mlp_in, nullptr, z_bf, H_);

  // 7) t1 = relu(z @ W_m1^T + b_m1) -> bf16
  gemm_bf16_kernel<<<dim3(H_ / 128, ROWS_ / 128), 256, 0, stream>>>(
      z_bf, wm1_bf, b_m1, nullptr, nullptr, t1_bf, ROWS_, H_, H_, 1);

  // 8) final = mlp_in + relu(t1 @ W_m2^T + b_m2) -> bf16 (residual fused)
  gemm_bf16_kernel<<<dim3(H_ / 128, ROWS_ / 128), 256, 0, stream>>>(
      t1_bf, wm2_bf, b_m2, mlp_in, nullptr, fin_bf, ROWS_, H_, H_, 1);

  // 9) scores_pre = final @ W_out^T + b_out -> f32 straight into d_out
  //    (W_out zero-padded to VPAD_ rows so all B-tile loads are unconditional)
  gemm_bf16_kernel<<<dim3(VPAD_ / 128, ROWS_ / 128), 256, 0, stream>>>(
      fin_bf, wout_bf, b_out, nullptr, scores, nullptr, ROWS_, V_, H_, 0);

  // 10) in-place LN + log_softmax over V
  ln_logsoftmax_kernel<<<ROWS_, 256, 0, stream>>>(scores, ln_out_g, ln_out_b, V_);
}
